// AttentionConvHead_76630806495622
// MI455X (gfx1250) — matrix operand order repaired and checked by reading, not compiled, not run on hardware
//
#include <hip/hip_runtime.h>

typedef __attribute__((ext_vector_type(16))) _Float16 v16h;
typedef __attribute__((ext_vector_type(8)))  float    v8f;

#define T_DIM 128
#define N_DIM 207
#define C_DIM 64
#define O_DIM 64
#define KSZ   7
#define PADT  3
#define TB    16              // output t rows per workgroup
#define NB    16              // n columns per workgroup
#define THALO (TB + KSZ - 1)  // 22 K/V rows incl. halo
#define TN    (T_DIM * N_DIM) // 26496 floats per channel per batch

// LDS: Qs f32 [64][16][16] (65536 B) | Ks f16 [64][22][16] (45056 B) | Vs f16 (45056 B)
#define LDS_Q_BYTES  (O_DIM * TB * NB * 4)
#define LDS_KV_BYTES (O_DIM * THALO * NB * 2)
#define LDS_BYTES    (LDS_Q_BYTES + 2 * LDS_KV_BYTES)

// One 16(o) x 16(n) D-tile of a projection: D = W(16x64) * X(64x16) + bias,
// as two v_wmma_f32_16x16x32_f16 with the C accumulator preloaded with bias.
// Returns the v8f accumulator (C/D layout: lane half g, VGPR r -> M = g*8 + r).
__device__ __forceinline__ v8f proj_tile(const float* __restrict__ xb,   // batch+t+n addressed, + c*TN strided
                                         const float* __restrict__ W,
                                         const float* __restrict__ bias,
                                         int otile, int m, int g)
{
    const float* wrow = W + (otile * 16 + m) * C_DIM;
    v16h a0, a1, b0, b1;
    #pragma unroll
    for (int h = 0; h < 16; ++h) {
        // ISA 16-bit A 16x32 layout: lane m = M row; VGPR/half -> K index map.
        int vv = h >> 1, pp = h & 1;
        int kk = (vv < 4 ? 2 * vv : 16 + 2 * (vv - 4)) + pp + g * 8;
        a0[h] = (_Float16)wrow[kk];
        a1[h] = (_Float16)wrow[kk + 32];
        // ISA 16-bit B 32x16 layout: same K map as A (h<8: g*8+h, else 16+g*8+h-8).
        int cb = (h < 8 ? h : h + 8) + g * 8;
        b0[h] = (_Float16)xb[(size_t)cb * TN];
        b1[h] = (_Float16)xb[(size_t)(cb + 32) * TN];
    }
    v8f c;
    #pragma unroll
    for (int r = 0; r < 8; ++r) c[r] = bias[otile * 16 + g * 8 + r];
    c = __builtin_amdgcn_wmma_f32_16x16x32_f16(false, a0, false, b0, (short)0, c, false, false);
    c = __builtin_amdgcn_wmma_f32_16x16x32_f16(false, a1, false, b1, (short)0, c, false, false);
    return c;
}

__global__ __launch_bounds__(256) void attn_conv_fused(
    const float* __restrict__ q, const float* __restrict__ k, const float* __restrict__ v,
    const float* __restrict__ Wq, const float* __restrict__ bq,
    const float* __restrict__ Wk, const float* __restrict__ bk,
    const float* __restrict__ Wv, const float* __restrict__ bv,
    float* __restrict__ out)
{
    extern __shared__ char smem[];
    float*    Qs = (float*)smem;
    _Float16* Ks = (_Float16*)(smem + LDS_Q_BYTES);
    _Float16* Vs = (_Float16*)(smem + LDS_Q_BYTES + LDS_KV_BYTES);

    const int tid  = threadIdx.x;
    const int lane = tid & 31;
    // Scalarize the wave id: identical for all lanes of a wave, so force it
    // into an SGPR -> scalar loop control, no EXEC save/restore near WMMA.
    const int wave = __builtin_amdgcn_readfirstlane(tid >> 5);
    const int n0 = blockIdx.x * NB;
    const int t0 = blockIdx.y * TB;
    const int b  = blockIdx.z;

    const int m = lane & 15;   // A tile row / B tile column
    const int g = lane >> 4;   // lane half-group (ISA K-split)

    // Branch-free clamped n (edge block n0=192 reaches 207; clamped lanes are
    // computed on duplicated data and never stored to global).
    int ng = n0 + m; if (ng > N_DIM - 1) ng = N_DIM - 1;

    const size_t batch_off = (size_t)b * C_DIM * TN;
    const float* qb = q + batch_off + ng;
    const float* kb = k + batch_off + ng;
    const float* vb = v + batch_off + ng;

    // ---- Phase 1a: Q projection (4 otiles x 16 t-rows = 64 jobs, 8 per wave) ----
    for (int j = wave; j < 64; j += 8) {
        const int otile = j >> 4, pt = j & 15;
        const int tg = t0 + pt;                       // always in [0,128)
        v8f c = proj_tile(qb + (size_t)tg * N_DIM, Wq, bq, otile, m, g);
        #pragma unroll
        for (int r = 0; r < 8; ++r)
            Qs[(otile * 16 + g * 8 + r) * (TB * NB) + pt * NB + m] = c[r];
    }

    // ---- Phase 1b/1c: K and V projections with +-3 t halo (4 x 22 = 88 jobs each) ----
    // Reference pads K,V with zeros AFTER projection: halo rows outside [0,T)
    // must be exact zeros (they still enter the softmax denominator).
    for (int j = wave; j < 88; j += 8) {
        const int otile = j / 22, pt = j % 22;
        const int tg = t0 - PADT + pt;
        const int tc = tg < 0 ? 0 : (tg > T_DIM - 1 ? T_DIM - 1 : tg);
        const float vmask = (tg >= 0 && tg < T_DIM) ? 1.0f : 0.0f;   // branch-free zero-fill
        v8f ck = proj_tile(kb + (size_t)tc * N_DIM, Wk, bk, otile, m, g);
        v8f cv = proj_tile(vb + (size_t)tc * N_DIM, Wv, bv, otile, m, g);
        #pragma unroll
        for (int r = 0; r < 8; ++r) {
            int off = (otile * 16 + g * 8 + r) * (THALO * NB) + pt * NB + m;
            Ks[off] = (_Float16)(ck[r] * vmask);
            Vs[off] = (_Float16)(cv[r] * vmask);
        }
    }

    __syncthreads();

    // ---- Phase 2: 7-tap windowed softmax from LDS (fp32 VALU) ----
    const int o   = tid >> 2;        // 0..63
    const int nb4 = (tid & 3) * 4;   // 4 n's per thread
    for (int dn = 0; dn < 4; ++dn) {
        const int n   = nb4 + dn;
        const int ngo = n0 + n;
        if (ngo >= N_DIM) continue;  // VALU-only divergence, no WMMA here
        const float*    qrow = Qs + o * (TB * NB) + n;
        const _Float16* krow = Ks + o * (THALO * NB) + n;
        const _Float16* vrow = Vs + o * (THALO * NB) + n;
        for (int t = 0; t < TB; ++t) {
            float qv = qrow[t * NB];
            float s[KSZ];
            float mx = -3.4e38f;
            #pragma unroll
            for (int i = 0; i < KSZ; ++i) {
                s[i] = qv * (float)krow[(t + i) * NB];
                mx = fmaxf(mx, s[i]);
            }
            float den = 0.f, num = 0.f;
            #pragma unroll
            for (int i = 0; i < KSZ; ++i) {
                float e = __expf(s[i] - mx);
                den += e;
                num += e * (float)vrow[(t + i) * NB];
            }
            out[((size_t)(b * O_DIM + o) * T_DIM + (t0 + t)) * N_DIM + ngo] = num / den;
        }
    }
}

extern "C" void kernel_launch(void* const* d_in, const int* in_sizes, int n_in,
                              void* d_out, int out_size, void* d_ws, size_t ws_size,
                              hipStream_t stream) {
    const float* q  = (const float*)d_in[0];
    const float* k  = (const float*)d_in[1];
    const float* v  = (const float*)d_in[2];
    const float* Wq = (const float*)d_in[3];
    const float* bq = (const float*)d_in[4];
    const float* Wk = (const float*)d_in[5];
    const float* bk = (const float*)d_in[6];
    const float* Wv = (const float*)d_in[7];
    const float* bv = (const float*)d_in[8];
    float* out = (float*)d_out;

    const int B = in_sizes[0] / (C_DIM * T_DIM * N_DIM);   // 8
    dim3 grid((N_DIM + NB - 1) / NB, T_DIM / TB, B);       // (13, 8, 8)
    dim3 block(256);
    attn_conv_fused<<<grid, block, LDS_BYTES, stream>>>(q, k, v, Wq, bq, Wk, bk, Wv, bv, out);
}
